// VQ_VAE_Segment_59691455480146
// MI455X (gfx1250) — matrix operand order repaired and checked by reading, not compiled, not run on hardware
//
#include <hip/hip_runtime.h>
#include <hip/hip_bf16.h>

typedef __attribute__((ext_vector_type(16))) _Float16 v16h;
typedef __attribute__((ext_vector_type(8)))  _Float16 v8h;
typedef __attribute__((ext_vector_type(8)))  float    v8f;
typedef __attribute__((ext_vector_type(4)))  float    v4f;

#define BB 256
#define TT 512
#define DD 96
#define HH 128
#define G4 512
#define LL 128
#define KK 1024
#define BT (BB * TT)  // 131072

// f16 packed-fragment weight region offsets (in halves)
#define OFF_ENCWIH 0        // 512*96   = 49152
#define OFF_ENCWHH 49152    // 512*128  = 65536
#define OFF_DECWIH 114688   // 512*128
#define OFF_DECWHH 180224   // 512*128
#define OFF_FCMU   245760   // 128*128  = 16384
#define OFF_FCLV   262144   // 128*128
#define OFF_DECFC  278528   // 96*128   = 12288
#define OFF_CB     290816   // 1024*128 = 131072

// -------- gfx1250 async-to-LDS path (guarded; fallback = register pipeline) --------
#if defined(__has_builtin)
#  if __has_builtin(__builtin_amdgcn_global_load_async_to_lds_b128) && \
      __has_builtin(__builtin_amdgcn_s_wait_asynccnt)
#    define HAVE_ASYNC_LDS 1
#  endif
#endif

#ifdef HAVE_ASYNC_LDS
typedef int v4i_t __attribute__((vector_size(16)));
typedef __attribute__((address_space(1))) v4i_t* gv4i_p;   // global AS pointer to int4
typedef __attribute__((address_space(3))) v4i_t* lv4i_p;   // LDS AS pointer to int4
__device__ static inline void async_copy16(const void* gsrc, void* ldst) {
  gv4i_p g = (gv4i_p)(unsigned long long)gsrc;             // int round-trip -> AS1 ptr
  lv4i_p l = (lv4i_p)(unsigned)(unsigned long long)ldst;   // low 32 bits = LDS offset
  __builtin_amdgcn_global_load_async_to_lds_b128(g, l, 0, 0);
}
#endif

// ---------------- WMMA fragment loaders ----------------
__device__ static inline v16h load_a_from_f32(const float* __restrict__ X, int ld,
                                              int m0, int k0, int lane) {
  const float* row = X + (size_t)(m0 + (lane & 15)) * ld + k0 + ((lane >> 4) << 3);
  v4f x0 = *(const v4f*)(row);
  v4f x1 = *(const v4f*)(row + 4);
  v4f x2 = *(const v4f*)(row + 16);
  v4f x3 = *(const v4f*)(row + 20);
  v16h a;
#pragma unroll
  for (int i = 0; i < 4; ++i) {
    a[i]      = (_Float16)x0[i];
    a[4 + i]  = (_Float16)x1[i];
    a[8 + i]  = (_Float16)x2[i];
    a[12 + i] = (_Float16)x3[i];
  }
  return a;
}

__device__ static inline v16h load_a_from_h16(const _Float16* __restrict__ X, int ld,
                                              int m0, int k0, int lane) {
  const _Float16* row = X + (size_t)(m0 + (lane & 15)) * ld + k0 + ((lane >> 4) << 3);
  v8h lo = *(const v8h*)row;
  v8h hi = *(const v8h*)(row + 16);
  v16h a;
#pragma unroll
  for (int i = 0; i < 8; ++i) { a[i] = lo[i]; a[8 + i] = hi[i]; }
  return a;
}

// Packed B: fragment frag=(n0/16)*(K/32)+(k0/32); entry (lane,i) = W[(n0+lane%16)*K + k0 + 8*(lane/16) + 16*(i/8) + i%8].
__device__ static inline v16h load_b_packed(const _Float16* __restrict__ Wp, int Kd,
                                            int n0, int k0, int lane) {
  const _Float16* p = Wp + ((((size_t)(n0 >> 4) * (size_t)(Kd >> 5)) + (size_t)(k0 >> 5)) << 9)
                        + ((size_t)lane << 4);
  return *(const v16h*)p;
}

// ---------------- prep ----------------
__device__ static inline void pack_b(const float* __restrict__ W, _Float16* __restrict__ dst,
                                     int N, int K, int gid, int gsz) {
  const int ktn = K >> 5;
  const int total = N * K;
  for (int e = gid; e < total; e += gsz) {
    int frag = e >> 9, within = e & 511;
    int lane = within >> 4, i = within & 15;
    int nt = frag / ktn, kt = frag - nt * ktn;
    dst[e] = (_Float16)W[(size_t)(nt * 16 + (lane & 15)) * K + kt * 32 + ((lane >> 4) << 3) + ((i >> 3) << 4) + (i & 7)];
  }
}

__global__ void prep_kernel(const float* encWih, const float* encWhh,
                            const float* decWih, const float* decWhh,
                            const float* fcMuW, const float* fcLvW,
                            const float* decFcW, const float* cb,
                            const float* encBih, const float* encBhh,
                            const float* decBih, const float* decBhh,
                            _Float16* w16, float* encB, float* decB,
                            float* cnorm, float* acc) {
  const int gid = blockIdx.x * blockDim.x + threadIdx.x;
  const int gsz = gridDim.x * blockDim.x;
  pack_b(encWih, w16 + OFF_ENCWIH, 512, 96, gid, gsz);
  pack_b(encWhh, w16 + OFF_ENCWHH, 512, 128, gid, gsz);
  pack_b(decWih, w16 + OFF_DECWIH, 512, 128, gid, gsz);
  pack_b(decWhh, w16 + OFF_DECWHH, 512, 128, gid, gsz);
  pack_b(fcMuW, w16 + OFF_FCMU, 128, 128, gid, gsz);
  pack_b(fcLvW, w16 + OFF_FCLV, 128, 128, gid, gsz);
  pack_b(decFcW, w16 + OFF_DECFC, 96, 128, gid, gsz);
  pack_b(cb, w16 + OFF_CB, 1024, 128, gid, gsz);
  for (int i = gid; i < 512; i += gsz) {
    encB[i] = encBih[i] + encBhh[i];
    decB[i] = decBih[i] + decBhh[i];
  }
  for (int i = gid; i < 1024; i += gsz) {
    float s = 0.f;
#pragma unroll 8
    for (int k = 0; k < 128; ++k) { float v = cb[i * 128 + k]; s += v * v; }
    cnorm[i] = s;
  }
  if (gid < 16) acc[gid] = 0.f;
}

// ---------------- generic GEMM: Out[M,N] = Xf32[M,K] @ Wpacked^T + bias ----------------
__global__ void gemm_xw(const float* __restrict__ X, const _Float16* __restrict__ Wp,
                        const float* __restrict__ bias, float* __restrict__ Out,
                        int M, int N, int Kd) {
  const int tid  = threadIdx.x;
  const int lane = tid & 31;
  const int wave = tid >> 5;
  const int m0 = blockIdx.y * 64 + (wave & 3) * 16;
  const int n0 = blockIdx.x * 32 + (wave >> 2) * 16;
  __builtin_prefetch(X + (size_t)(m0 + (lane & 15)) * Kd + ((lane >> 4) * (Kd >> 1)), 0, 1);
  v8f c = {};
  for (int k0 = 0; k0 < Kd; k0 += 32) {
    v16h a = load_a_from_f32(X, Kd, m0, k0, lane);
    v16h b = load_b_packed(Wp, Kd, n0, k0, lane);
    c = __builtin_amdgcn_wmma_f32_16x16x32_f16(false, a, false, b, (short)0, c, false, false);
  }
  const int n = n0 + (lane & 15);
  const float bv = bias[n];
#pragma unroll
  for (int r = 0; r < 8; ++r) {
    int m = m0 + r + ((lane >> 4) << 3);
    Out[(size_t)m * N + n] = c[r] + bv;
  }
}

// ---------------- LSTM recurrence: 16 batch rows per WG, loop over T ----------------
// Whh fragments register-resident; next-step px prefetched via async-to-LDS (or
// register software pipeline when the async builtins are unavailable).
__global__ void lstm_rec(const float* __restrict__ preX, const _Float16* __restrict__ WhhP,
                         float* __restrict__ hout) {
  __shared__ _Float16 hF16[16 * 128];   // 4 KB
  __shared__ float    gates[16 * 512];  // 32 KB
  __shared__ float    cSt[16 * 128];    // 8 KB
#ifdef HAVE_ASYNC_LDS
  __shared__ float    pxLds[2][16 * 512]; // 64 KB double buffer
#endif
  const int tid  = threadIdx.x;
  const int lane = tid & 31;
  const int wave = tid >> 5;
  const int b0 = blockIdx.x * 16;
  for (int p = tid; p < 2048; p += 256) { hF16[p] = (_Float16)0.f; cSt[p] = 0.f; }

  // register-resident Whh fragments: bfr[j][kk] for n0 = wave*64 + j*16, k0 = kk*32
  v16h bfr[4][4];
#pragma unroll
  for (int j = 0; j < 4; ++j)
#pragma unroll
    for (int kk = 0; kk < 4; ++kk)
      bfr[j][kk] = load_b_packed(WhhP, 128, wave * 64 + j * 16, kk * 32, lane);

#ifdef HAVE_ASYNC_LDS
  // prologue: async-stage px tile for t=0 (2048 x 16B chunks, 8 per thread)
#pragma unroll
  for (int q = 0; q < 8; ++q) {
    int p = tid * 8 + q;                 // 0..2047
    int row = p >> 7, cidx = (p & 127) << 2;
    async_copy16(&preX[(size_t)((b0 + row) * TT + 0) * 512 + cidx],
                 &pxLds[0][row * 512 + cidx]);
  }
  __builtin_amdgcn_s_wait_asynccnt(0);
#else
  v8f pxr[4];
#pragma unroll
  for (int j = 0; j < 4; ++j) {
    const int n = wave * 64 + j * 16 + (lane & 15);
#pragma unroll
    for (int r = 0; r < 8; ++r)
      pxr[j][r] = preX[(size_t)((b0 + r + ((lane >> 4) << 3)) * TT + 0) * 512 + n];
  }
#endif
  __syncthreads();

  for (int t = 0; t < TT; ++t) {
#ifdef HAVE_ASYNC_LDS
    const int buf = t & 1;
    // kick off async staging of t+1 while we compute t
    if (t + 1 < TT) {
#pragma unroll
      for (int q = 0; q < 8; ++q) {
        int p = tid * 8 + q;
        int row = p >> 7, cidx = (p & 127) << 2;
        async_copy16(&preX[(size_t)((b0 + row) * TT + (t + 1)) * 512 + cidx],
                     &pxLds[buf ^ 1][row * 512 + cidx]);
      }
    }
#endif
    // ---- gate GEMM: each wave owns 4 N-tiles of 16 cols ----
#pragma unroll
    for (int j = 0; j < 4; ++j) {
      const int n0 = wave * 64 + j * 16;
      const int n  = n0 + (lane & 15);
      v8f c;
#ifdef HAVE_ASYNC_LDS
#pragma unroll
      for (int r = 0; r < 8; ++r)
        c[r] = pxLds[buf][(r + ((lane >> 4) << 3)) * 512 + n];
#else
      c = pxr[j];
#endif
#pragma unroll
      for (int kk = 0; kk < 4; ++kk) {
        v16h a = load_a_from_h16(hF16, 128, 0, kk * 32, lane);
        c = __builtin_amdgcn_wmma_f32_16x16x32_f16(false, a, false, bfr[j][kk], (short)0, c,
                                                   false, false);
      }
#pragma unroll
      for (int r = 0; r < 8; ++r)
        gates[(r + ((lane >> 4) << 3)) * 512 + n] = c[r];
#ifndef HAVE_ASYNC_LDS
      // software pipeline: fetch next step's px for this tile behind the WMMAs
      if (t + 1 < TT) {
#pragma unroll
        for (int r = 0; r < 8; ++r)
          pxr[j][r] = preX[(size_t)((b0 + r + ((lane >> 4) << 3)) * TT + (t + 1)) * 512 + n];
      }
#endif
    }
    __syncthreads();
    // ---- elementwise cell update: 8 (row,j) pairs per thread ----
#pragma unroll
    for (int q = 0; q < 8; ++q) {
      int p = tid * 8 + q;              // 0..2047
      int row = p >> 7, jj = p & 127;
      float gi = gates[row * 512 + jj];
      float gf = gates[row * 512 + 128 + jj];
      float gg = gates[row * 512 + 256 + jj];
      float go = gates[row * 512 + 384 + jj];
      float iv = 1.f / (1.f + __expf(-gi));
      float fv = 1.f / (1.f + __expf(-gf));
      float gv = tanhf(gg);
      float ov = 1.f / (1.f + __expf(-go));
      float cv = fv * cSt[row * 128 + jj] + iv * gv;
      cSt[row * 128 + jj] = cv;
      float hv = ov * tanhf(cv);
      hF16[row * 128 + jj] = (_Float16)hv;
      hout[(size_t)((b0 + row) * TT + t) * 128 + jj] = hv;
    }
#ifdef HAVE_ASYNC_LDS
    __builtin_amdgcn_s_wait_asynccnt(0);  // drain t+1 staging before next step reads it
#endif
    __syncthreads();
  }
}

// ---------------- z = mu + eps*exp(0.5*lv), KL accumulation ----------------
__global__ void z_kl_kernel(const float* __restrict__ mu, const float* __restrict__ lv,
                            const float* __restrict__ eps, float* __restrict__ z,
                            float* __restrict__ acc) {
  const int i = blockIdx.x * 256 + threadIdx.x;   // exact cover: BT*128
  float muv = mu[i], lvv = lv[i];
  float ev = __expf(lvv);
  z[i] = muv + eps[i] * __expf(0.5f * lvv);
  float kl = -0.5f * (1.f + lvv - muv * muv - ev);
  for (int o = 16; o > 0; o >>= 1) kl += __shfl_down(kl, o, 32);
  __shared__ float red[8];
  const int lane = threadIdx.x & 31, wave = threadIdx.x >> 5;
  if (lane == 0) red[wave] = kl;
  __syncthreads();
  if (threadIdx.x == 0) {
    float s = 0.f;
    for (int w = 0; w < 8; ++w) s += red[w];
    atomicAdd(acc, s);
  }
}

// ---------------- VQ: WMMA z@cb^T, argmin via ds_min_u64, gather + loss ----------------
__global__ void vq_kernel(const float* __restrict__ z, const _Float16* __restrict__ cb16p,
                          const float* __restrict__ cnorm, const float* __restrict__ cb32,
                          float* __restrict__ q, float* __restrict__ idx_out,
                          float* __restrict__ acc) {
  __shared__ _Float16 zF16[16 * 128];
  __shared__ unsigned long long best[16];
  __shared__ float red[8];
  const int tid  = threadIdx.x;
  const int lane = tid & 31;
  const int wave = tid >> 5;
  const int row0 = blockIdx.x * 16;
  for (int p = tid; p < 2048; p += 256) zF16[p] = (_Float16)z[(size_t)row0 * 128 + p];
  if (tid < 16) best[tid] = 0xFFFFFFFFFFFFFFFFull;
  __syncthreads();
  v16h afr[4];
#pragma unroll
  for (int kk = 0; kk < 4; ++kk) afr[kk] = load_a_from_h16(zF16, 128, 0, kk * 32, lane);
  for (int pass = 0; pass < 8; ++pass) {
    const int n0 = pass * 128 + wave * 16;
    v8f c = {};
#pragma unroll
    for (int kk = 0; kk < 4; ++kk) {
      v16h b = load_b_packed(cb16p, 128, n0, kk * 32, lane);
      c = __builtin_amdgcn_wmma_f32_16x16x32_f16(false, afr[kk], false, b, (short)0, c,
                                                 false, false);
    }
    const int n = n0 + (lane & 15);
    const float cn = cnorm[n];
#pragma unroll
    for (int r = 0; r < 8; ++r) {
      float d = cn - 2.f * c[r];   // ||z||^2 constant per row: drop for argmin
      unsigned kb = __float_as_uint(d);
      kb = (kb & 0x80000000u) ? ~kb : (kb | 0x80000000u);  // order-preserving key
      unsigned long long key = ((unsigned long long)kb << 32) | (unsigned)n;
      atomicMin(&best[r + ((lane >> 4) << 3)], key);       // ds_min_u64
    }
  }
  __syncthreads();
  if (tid < 16) idx_out[row0 + tid] = (float)(unsigned)(best[tid] & 0xFFFFFFFFull);
  float local = 0.f;
  for (int p = tid; p < 2048; p += 256) {
    int row = p >> 7, col = p & 127;
    int ci = (int)(unsigned)(best[row] & 0xFFFFFFFFull);
    float qv = cb32[(size_t)ci * 128 + col];
    float zv = z[(size_t)(row0 + row) * 128 + col];
    float dd = qv - zv;
    local += dd * dd;
    q[(size_t)(row0 + row) * 128 + col] = qv;
  }
  for (int o = 16; o > 0; o >>= 1) local += __shfl_down(local, o, 32);
  if (lane == 0) red[wave] = local;
  __syncthreads();
  if (tid == 0) {
    float s = 0.f;
    for (int w = 0; w < 8; ++w) s += red[w];
    atomicAdd(acc + 1, s);
  }
}

// ---------------- final loss combine ----------------
__global__ void finalize_kernel(const float* acc, float* dst) {
  dst[0] = acc[1] * (1.25f / ((float)BT * 128.f)) + acc[0];
}

extern "C" void kernel_launch(void* const* d_in, const int* in_sizes, int n_in,
                              void* d_out, int out_size, void* d_ws, size_t ws_size,
                              hipStream_t stream) {
  const float* traj   = (const float*)d_in[0];
  const float* eps    = (const float*)d_in[1];
  const float* encWih = (const float*)d_in[2];
  const float* encWhh = (const float*)d_in[3];
  const float* encBih = (const float*)d_in[4];
  const float* encBhh = (const float*)d_in[5];
  const float* fcMuW  = (const float*)d_in[6];
  const float* fcMuB  = (const float*)d_in[7];
  const float* fcLvW  = (const float*)d_in[8];
  const float* fcLvB  = (const float*)d_in[9];
  const float* cb     = (const float*)d_in[10];
  const float* decWih = (const float*)d_in[11];
  const float* decWhh = (const float*)d_in[12];
  const float* decBih = (const float*)d_in[13];
  const float* decBhh = (const float*)d_in[14];
  const float* decFcW = (const float*)d_in[15];
  const float* decFcB = (const float*)d_in[16];

  // workspace layout
  float* preX  = (float*)d_ws;                    // BT*512
  float* hbuf  = preX + (size_t)BT * 512;         // BT*128 (enc h, reused as dec h)
  float* qbuf  = hbuf + (size_t)BT * 128;         // BT*128 quantized
  float* cnorm = qbuf + (size_t)BT * 128;         // 1024
  float* encB  = cnorm + 1024;                    // 512
  float* decB  = encB + 512;                      // 512
  float* acc   = decB + 512;                      // 16 (acc[0]=KL, acc[1]=VQ)
  _Float16* w16 = (_Float16*)(acc + 16);          // packed f16 fragments
  // preX region reuse between encoder and decoder phases:
  float* mu   = preX;
  float* lv   = preX + (size_t)BT * 128;
  float* zbuf = preX + (size_t)BT * 256;

  float* out   = (float*)d_out;
  float* recon = out;                             // BT*96
  float* lossp = out + (size_t)BT * 96;           // 1
  float* idxp  = lossp + 1;                       // BT

  prep_kernel<<<dim3(1024), dim3(256), 0, stream>>>(
      encWih, encWhh, decWih, decWhh, fcMuW, fcLvW, decFcW, cb,
      encBih, encBhh, decBih, decBhh, w16, encB, decB, cnorm, acc);

  gemm_xw<<<dim3(16, 2048), 256, 0, stream>>>(traj, w16 + OFF_ENCWIH, encB, preX, BT, 512, 96);
  lstm_rec<<<dim3(16), 256, 0, stream>>>(preX, w16 + OFF_ENCWHH, hbuf);

  gemm_xw<<<dim3(4, 2048), 256, 0, stream>>>(hbuf, w16 + OFF_FCMU, fcMuB, mu, BT, 128, 128);
  gemm_xw<<<dim3(4, 2048), 256, 0, stream>>>(hbuf, w16 + OFF_FCLV, fcLvB, lv, BT, 128, 128);
  z_kl_kernel<<<dim3((BT * 128) / 256), 256, 0, stream>>>(mu, lv, eps, zbuf, acc);

  vq_kernel<<<dim3(BT / 16), 256, 0, stream>>>(zbuf, w16 + OFF_CB, cnorm, cb, qbuf, idxp, acc);

  gemm_xw<<<dim3(16, 2048), 256, 0, stream>>>(qbuf, w16 + OFF_DECWIH, decB, preX, BT, 512, 128);
  lstm_rec<<<dim3(16), 256, 0, stream>>>(preX, w16 + OFF_DECWHH, hbuf);
  gemm_xw<<<dim3(3, 2048), 256, 0, stream>>>(hbuf, w16 + OFF_DECFC, decFcB, recon, BT, 96, 128);

  finalize_kernel<<<1, 1, 0, stream>>>(acc, lossp);
}
